// PriorAwareIPR_MPNN_67654324846758
// MI455X (gfx1250) — matrix operand order repaired and checked by reading, not compiled, not run on hardware
//
#include <hip/hip_runtime.h>

// ---------------- problem constants ----------------
#define NN 50000
#define HH 256
#define VV 4
#define BB 64
#define NT 16          // node tile (rows per block in GEMM kernels)
#define NBLK (NN / NT) // 3125
#define ASTRIDE 272    // padded LDS row stride in halves (16B aligned chunks)

typedef _Float16 vhalf8  __attribute__((ext_vector_type(8)));
typedef _Float16 vhalf16 __attribute__((ext_vector_type(16)));
typedef float    vfloat8 __attribute__((ext_vector_type(8)));
typedef int      gvec4   __attribute__((vector_size(16)));   // matches builtin param type

// ---------------- CDNA5 async global->LDS staging (guarded) ----------------
#if __has_builtin(__builtin_amdgcn_global_load_async_to_lds_b128) && \
    __has_builtin(__builtin_amdgcn_s_wait_asynccnt)
#define ASYNC_OK 1
#endif

// copy 16 bytes global -> LDS
__device__ __forceinline__ void stage16B(const _Float16* g, _Float16* l) {
#ifdef ASYNC_OK
    __builtin_amdgcn_global_load_async_to_lds_b128(
        (__attribute__((address_space(1))) gvec4*)(_Float16*)g,
        (__attribute__((address_space(3))) gvec4*)l, 0, 0);
#else
    *(vhalf8*)l = *(const vhalf8*)g;
#endif
}

__device__ __forceinline__ void async_join() {
#ifdef ASYNC_OK
    __builtin_amdgcn_s_wait_asynccnt(0);
#endif
}

// ---------------- WMMA helpers ----------------
__device__ __forceinline__ vhalf16 frag_combine(vhalf8 lo, vhalf8 hi) {
    vhalf16 a;
#pragma unroll
    for (int i = 0; i < 8; ++i) { a[i] = lo[i]; a[i + 8] = hi[i]; }
    return a;
}

// A fragment (16x32 f16, M=node rows) from LDS tile, row stride ASTRIDE halves.
// lane L: row m=L&15; K halves [kb..kb+7] and [16+kb..16+kb+7], kb=(L>>4)*8.
__device__ __forceinline__ vhalf16 loadA_lds(const _Float16* sA, int lane, int k0) {
    int m = lane & 15, kb = (lane >> 4) * 8;
    const _Float16* p = sA + m * ASTRIDE + k0 + kb;
    vhalf8 lo = *(const vhalf8*)p;
    vhalf8 hi = *(const vhalf8*)(p + 16);
    return frag_combine(lo, hi);
}

// B fragment (32x16 f16, K x outfeat) from half weights W[M,256] row-major.
// lane L: out col n0+(L&15); K halves [kh..kh+7] and [16+kh..16+kh+7], kh=(L>>4)*8.
__device__ __forceinline__ vhalf16 loadB_glb(const _Float16* W, int n0, int lane, int k0) {
    int n = lane & 15, kh = (lane >> 4) * 8;
    const _Float16* p = W + (size_t)(n0 + n) * HH + k0 + kh;
    vhalf8 lo = *(const vhalf8*)p;
    vhalf8 hi = *(const vhalf8*)(p + 16);
    return frag_combine(lo, hi);
}

__device__ __forceinline__ vfloat8 wmma16(vhalf16 a, vhalf16 b, vfloat8 c) {
    return __builtin_amdgcn_wmma_f32_16x16x32_f16(false, a, false, b, (short)0, c, false, false);
}

__device__ __forceinline__ float sigm(float x) { return 1.0f / (1.0f + expf(-x)); }

// ---------------- conversion kernels ----------------
__global__ void f2h_kernel(const float* __restrict__ src, _Float16* __restrict__ dst, int n) {
    int i = blockIdx.x * blockDim.x + threadIdx.x;
    if (i < n) dst[i] = (_Float16)src[i];
}

// rW1 is [256, 257]; keep first 256 columns of each row.
__global__ void rw1_conv_kernel(const float* __restrict__ rW1, _Float16* __restrict__ dst) {
    int i = blockIdx.x * blockDim.x + threadIdx.x; // 0..65535
    int row = i >> 8, col = i & 255;
    dst[i] = (_Float16)rW1[row * 257 + col];
}

// ---------------- kernel 1: router MLP + gumbel top-k + r2v ----------------
__global__ __launch_bounds__(256) void k1_router_r2v(
    const _Float16* __restrict__ X16, const float* __restrict__ priors,
    const float* __restrict__ gumbel, const float* __restrict__ rW1f,
    const float* __restrict__ rb1, const float* __restrict__ rW2,
    const float* __restrict__ rb2, const float* __restrict__ mb,
    const _Float16* __restrict__ rW1h, const _Float16* __restrict__ mWh,
    float* __restrict__ w_out, float* __restrict__ r2v_out) {

    __shared__ _Float16 sA[NT][ASTRIDE];
    __shared__ float sH1[NT][HH];
    __shared__ float sLast[HH];
    __shared__ float sLogits[NT][VV];

    const int t = threadIdx.x;
    const int node0 = blockIdx.x * NT;
    const int wv = t >> 5, lane = t & 31;

    // stage A tile (16 x 256 halves) into LDS via async copy (32B per thread)
    {
        int row = t >> 4, seg = t & 15;
        const _Float16* g = X16 + (size_t)(node0 + row) * HH + seg * 16;
        _Float16* l = &sA[row][seg * 16];
        stage16B(g, l);
        stage16B(g + 8, l + 8);
    }
    sLast[t] = rW1f[t * 257 + 256]; // rW1[:,256]
    async_join();
    __syncthreads();

    // 4 output tiles per wave sharing one A fragment per k-step:
    //   acc0: router cols wv*16       acc1: router cols (wv+8)*16
    //   acc2: r2v    cols wv*16       acc3: r2v    cols (wv+8)*16
    const int n0a = wv * 16, n0b = (wv + 8) * 16;
    vfloat8 acc0 = {}, acc1 = {}, acc2 = {}, acc3 = {};
#pragma unroll
    for (int kk = 0; kk < 8; ++kk) {
        int k0 = kk * 32;
        vhalf16 a  = loadA_lds(&sA[0][0], lane, k0);
        vhalf16 b0 = loadB_glb(rW1h, n0a, lane, k0);
        vhalf16 b1 = loadB_glb(rW1h, n0b, lane, k0);
        vhalf16 b2 = loadB_glb(mWh, n0a, lane, k0);
        vhalf16 b3 = loadB_glb(mWh, n0b, lane, k0);
        acc0 = wmma16(a, b0, acc0);
        acc1 = wmma16(a, b1, acc1);
        acc2 = wmma16(a, b2, acc2);
        acc3 = wmma16(a, b3, acc3);
    }

    // per-lane row set: m = r + 8*(lane>>4)
    float pr[8];
#pragma unroll
    for (int r = 0; r < 8; ++r) pr[r] = priors[node0 + r + 8 * (lane >> 4)];

    {   // router tiles -> relu(h1) in LDS (bias + prior * rW1[:,256])
        int na = n0a + (lane & 15), nb = n0b + (lane & 15);
        float ba = rb1[na], la = sLast[na];
        float bb2 = rb1[nb], lb = sLast[nb];
#pragma unroll
        for (int r = 0; r < 8; ++r) {
            int m = r + 8 * (lane >> 4);
            float va = acc0[r] + ba + pr[r] * la;
            float vb = acc1[r] + bb2 + pr[r] * lb;
            sH1[m][na] = va > 0.0f ? va : 0.0f;
            sH1[m][nb] = vb > 0.0f ? vb : 0.0f;
        }
    }
    {   // r2v tiles -> global (+mb)
        int na = n0a + (lane & 15), nb = n0b + (lane & 15);
        float ba = mb[na], bb2 = mb[nb];
#pragma unroll
        for (int r = 0; r < 8; ++r) {
            int m = r + 8 * (lane >> 4);
            size_t row = (size_t)(node0 + m) * HH;
            r2v_out[row + na] = acc2[r] + ba;
            r2v_out[row + nb] = acc3[r] + bb2;
        }
    }
    __syncthreads();

    // logits = relu_h1 @ rW2.T + rb2   (V=4, VALU)
    if (t < NT * VV) {
        int nd = t >> 2, v = t & 3;
        float s = rb2[v];
        const float* wr = rW2 + v * HH;
        for (int k = 0; k < HH; ++k) s += sH1[nd][k] * wr[k];
        sLogits[nd][v] = s;
    }
    __syncthreads();

    // gumbel straight-through top-3-of-4: forward value = (hard - soft) + soft
    if (t < NT) {
        int gnode = node0 + t;
        float y[VV];
#pragma unroll
        for (int v = 0; v < VV; ++v) y[v] = sLogits[t][v] + gumbel[gnode * VV + v];
        int mi = 0; float mv = y[0];
#pragma unroll
        for (int v = 1; v < VV; ++v) if (y[v] <= mv) { mv = y[v]; mi = v; }
        float mx = y[0];
#pragma unroll
        for (int v = 1; v < VV; ++v) mx = y[v] > mx ? y[v] : mx;
        float e[VV], sum = 0.0f;
#pragma unroll
        for (int v = 0; v < VV; ++v) { e[v] = expf((y[v] - mx) * 2.0f); sum += e[v]; } // 1/TEMP = 2
        float inv = 1.0f / sum;
#pragma unroll
        for (int v = 0; v < VV; ++v) {
            float soft = e[v] * inv;
            float hard = (v == mi) ? 0.0f : 1.0f;
            w_out[gnode * VV + v] = (hard - soft) + soft;
        }
    }
}

// ---------------- kernel 2: segment pooling + VN self-attention ----------------
__device__ __forceinline__ int lower_bound(const int* a, int n, int key) {
    int lo = 0, hi = n;
    while (lo < hi) { int mid = (lo + hi) >> 1; if (a[mid] < key) lo = mid + 1; else hi = mid; }
    return lo;
}

__global__ __launch_bounds__(256) void k2_pool_attn(
    const float* __restrict__ r2v, const float* __restrict__ w,
    const int* __restrict__ batch, const float* __restrict__ vn_init,
    const float* __restrict__ ipw, const float* __restrict__ ipb,
    const float* __restrict__ ow, const float* __restrict__ ob,
    float* __restrict__ vn_out) {

    __shared__ float sWrow[64][VV];
    __shared__ float sVN[VV][HH];
    __shared__ float sQKV[VV][3 * HH];
    __shared__ float sS[VV][4][VV];   // [q][head][kv]
    __shared__ float sO[VV][HH];

    const int t = threadIdx.x;
    const int b = blockIdx.x;
    const int start = lower_bound(batch, NN, b);
    const int end = lower_bound(batch, NN, b + 1);

    float acc0 = 0.f, acc1 = 0.f, acc2 = 0.f, acc3 = 0.f;
    for (int base = start; base < end; base += 64) {
        int cnt = end - base; if (cnt > 64) cnt = 64;
        int row = t >> 2, col = t & 3;
        if (row < cnt) sWrow[row][col] = w[(size_t)(base + row) * VV + col];
        __syncthreads();
        for (int j = 0; j < cnt; ++j) {
            float rv = r2v[(size_t)(base + j) * HH + t];
            acc0 += rv * sWrow[j][0];
            acc1 += rv * sWrow[j][1];
            acc2 += rv * sWrow[j][2];
            acc3 += rv * sWrow[j][3];
        }
        __syncthreads();
    }
    sVN[0][t] = vn_init[0 * HH + t] + acc0;
    sVN[1][t] = vn_init[1 * HH + t] + acc1;
    sVN[2][t] = vn_init[2 * HH + t] + acc2;
    sVN[3][t] = vn_init[3 * HH + t] + acc3;
    __syncthreads();

    // qkv = vn @ ipw.T + ipb  (4 x 768 outputs)
    for (int o = t; o < VV * 3 * HH; o += 256) {
        int v = o / (3 * HH), m = o - v * (3 * HH);
        float s = ipb[m];
        const float* wr = ipw + (size_t)m * HH;
        for (int k = 0; k < HH; ++k) s += sVN[v][k] * wr[k];
        sQKV[v][m] = s;
    }
    __syncthreads();

    // scores: [q][head][kv], head dim 64, scale 1/8
    if (t < 64) {
        int q = (t >> 4) & 3, hd = (t >> 2) & 3, kv = t & 3;
        float s = 0.f;
        for (int d = 0; d < 64; ++d)
            s += sQKV[q][hd * 64 + d] * sQKV[kv][HH + hd * 64 + d];
        sS[q][hd][kv] = s * 0.125f;
    }
    __syncthreads();
    if (t < 16) {
        int q = t >> 2, hd = t & 3;
        float* p = &sS[q][hd][0];
        float mx = p[0];
        for (int i = 1; i < 4; ++i) mx = p[i] > mx ? p[i] : mx;
        float e0 = expf(p[0] - mx), e1 = expf(p[1] - mx), e2 = expf(p[2] - mx), e3 = expf(p[3] - mx);
        float inv = 1.0f / (e0 + e1 + e2 + e3);
        p[0] = e0 * inv; p[1] = e1 * inv; p[2] = e2 * inv; p[3] = e3 * inv;
    }
    __syncthreads();
    // o = attn @ v
    {
        int hd = t >> 6;
#pragma unroll
        for (int q = 0; q < VV; ++q) {
            float s = 0.f;
#pragma unroll
            for (int kv = 0; kv < VV; ++kv) s += sS[q][hd][kv] * sQKV[kv][2 * HH + t];
            sO[q][t] = s;
        }
    }
    __syncthreads();
    // vn += o @ ow.T + ob ; write out
    {
        const float* wr = ow + (size_t)t * HH;
#pragma unroll
        for (int q = 0; q < VV; ++q) {
            float s = ob[t];
            for (int k = 0; k < HH; ++k) s += sO[q][k] * wr[k];
            vn_out[(size_t)b * (VV * HH) + q * HH + t] = sVN[q][t] + s;
        }
    }
}

// ---------------- kernel 3: vn->real gather + GRU via 6-way WMMA ----------------
__global__ __launch_bounds__(256) void k3_gru(
    const _Float16* __restrict__ X16, const float* __restrict__ real_nodes,
    const float* __restrict__ w, const int* __restrict__ batch,
    const float* __restrict__ vn, const _Float16* __restrict__ wihH,
    const _Float16* __restrict__ whhH, const float* __restrict__ bih,
    const float* __restrict__ bhh, float* __restrict__ outp) {

    __shared__ _Float16 sMsg[NT][ASTRIDE];
    __shared__ _Float16 sX[NT][ASTRIDE];
    __shared__ float sWn[NT][VV];
    __shared__ int sB[NT];

    const int t = threadIdx.x;
    const int node0 = blockIdx.x * NT;
    const int wv = t >> 5, lane = t & 31;

    // stage X tile into LDS via async copy
    {
        int row = t >> 4, seg = t & 15;
        const _Float16* g = X16 + (size_t)(node0 + row) * HH + seg * 16;
        _Float16* l = &sX[row][seg * 16];
        stage16B(g, l);
        stage16B(g + 8, l + 8);
    }
    if (t < NT) sB[t] = batch[node0 + t];
    if (t < NT * VV) sWn[t >> 2][t & 3] = w[(size_t)(node0 + (t >> 2)) * VV + (t & 3)];
    __syncthreads();

    // real_msg[nd][h] = sum_v vn[batch[nd]][v][h] * w[nd][v]  -> f16 LDS
#pragma unroll 4
    for (int nd = 0; nd < NT; ++nd) {
        const float* vrow = vn + (size_t)sB[nd] * (VV * HH);
        float m = vrow[t] * sWn[nd][0] + vrow[HH + t] * sWn[nd][1] +
                  vrow[2 * HH + t] * sWn[nd][2] + vrow[3 * HH + t] * sWn[nd][3];
        sMsg[nd][t] = (_Float16)m;
    }
    async_join();
    __syncthreads();

#pragma unroll
    for (int ti = 0; ti < 2; ++ti) {
        int h0 = (wv + ti * 8) * 16;
        vfloat8 agr = {}, agz = {}, agn = {}, ahr = {}, ahz = {}, ahn = {};
#pragma unroll
        for (int kk = 0; kk < 8; ++kk) {
            int k0 = kk * 32;
            vhalf16 am = loadA_lds(&sMsg[0][0], lane, k0);
            vhalf16 ax = loadA_lds(&sX[0][0], lane, k0);
            vhalf16 b0 = loadB_glb(wihH, h0, lane, k0);
            vhalf16 b1 = loadB_glb(wihH, h0 + HH, lane, k0);
            vhalf16 b2 = loadB_glb(wihH, h0 + 2 * HH, lane, k0);
            vhalf16 b3 = loadB_glb(whhH, h0, lane, k0);
            vhalf16 b4 = loadB_glb(whhH, h0 + HH, lane, k0);
            vhalf16 b5 = loadB_glb(whhH, h0 + 2 * HH, lane, k0);
            agr = wmma16(am, b0, agr);
            agz = wmma16(am, b1, agz);
            agn = wmma16(am, b2, agn);
            ahr = wmma16(ax, b3, ahr);
            ahz = wmma16(ax, b4, ahz);
            ahn = wmma16(ax, b5, ahn);
        }
        int n = h0 + (lane & 15);
        float bir = bih[n], biz = bih[HH + n], bin_ = bih[2 * HH + n];
        float bhr = bhh[n], bhz = bhh[HH + n], bhn = bhh[2 * HH + n];
#pragma unroll
        for (int r = 0; r < 8; ++r) {
            int m = r + 8 * (lane >> 4);
            size_t row = (size_t)(node0 + m) * HH;
            float x = real_nodes[row + n];
            float rr = sigm(agr[r] + bir + ahr[r] + bhr);
            float zz = sigm(agz[r] + biz + ahz[r] + bhz);
            float nn = tanhf(agn[r] + bin_ + rr * (ahn[r] + bhn));
            outp[row + n] = (1.0f - zz) * nn + zz * x;
        }
    }
}

// ---------------- launcher ----------------
extern "C" void kernel_launch(void* const* d_in, const int* in_sizes, int n_in,
                              void* d_out, int out_size, void* d_ws, size_t ws_size,
                              hipStream_t stream) {
    const float* real_nodes = (const float*)d_in[0];
    const float* priors     = (const float*)d_in[1];
    const float* gumbel     = (const float*)d_in[2];
    const int*   batch      = (const int*)d_in[3];
    const float* vn_init    = (const float*)d_in[4];
    const float* rW1        = (const float*)d_in[5];
    const float* rb1        = (const float*)d_in[6];
    const float* rW2        = (const float*)d_in[7];
    const float* rb2        = (const float*)d_in[8];
    const float* mW         = (const float*)d_in[9];
    const float* mb         = (const float*)d_in[10];
    const float* ipw        = (const float*)d_in[11];
    const float* ipb        = (const float*)d_in[12];
    const float* ow         = (const float*)d_in[13];
    const float* ob         = (const float*)d_in[14];
    const float* wih        = (const float*)d_in[15];
    const float* whh        = (const float*)d_in[16];
    const float* bih        = (const float*)d_in[17];
    const float* bhh        = (const float*)d_in[18];
    float* out = (float*)d_out;

    // workspace layout (all offsets multiples of 256 bytes)
    char* ws = (char*)d_ws;
    size_t off = 0;
    _Float16* X16  = (_Float16*)(ws + off); off += (size_t)NN * HH * 2;      // 25.6 MB
    _Float16* rW1h = (_Float16*)(ws + off); off += (size_t)HH * HH * 2;      // 128 KB
    _Float16* mWh  = (_Float16*)(ws + off); off += (size_t)HH * HH * 2;      // 128 KB
    _Float16* wihH = (_Float16*)(ws + off); off += (size_t)3 * HH * HH * 2;  // 384 KB
    _Float16* whhH = (_Float16*)(ws + off); off += (size_t)3 * HH * HH * 2;  // 384 KB
    float* wbuf    = (float*)(ws + off);    off += (size_t)NN * VV * 4;      // 800 KB
    float* r2vbuf  = (float*)(ws + off);    off += (size_t)NN * HH * 4;      // 51.2 MB
    float* vnbuf   = (float*)(ws + off);    off += (size_t)BB * VV * HH * 4; // 256 KB

    // f32 -> f16 staging
    f2h_kernel<<<(NN * HH + 255) / 256, 256, 0, stream>>>(real_nodes, X16, NN * HH);
    rw1_conv_kernel<<<(HH * HH) / 256, 256, 0, stream>>>(rW1, rW1h);
    f2h_kernel<<<(HH * HH) / 256, 256, 0, stream>>>(mW, mWh, HH * HH);
    f2h_kernel<<<(3 * HH * HH) / 256, 256, 0, stream>>>(wih, wihH, 3 * HH * HH);
    f2h_kernel<<<(3 * HH * HH) / 256, 256, 0, stream>>>(whh, whhH, 3 * HH * HH);

    k1_router_r2v<<<NBLK, 256, 0, stream>>>(X16, priors, gumbel, rW1, rb1, rW2, rb2,
                                            mb, rW1h, mWh, wbuf, r2vbuf);
    k2_pool_attn<<<BB, 256, 0, stream>>>(r2vbuf, wbuf, batch, vn_init, ipw, ipb,
                                         ow, ob, vnbuf);
    k3_gru<<<NBLK, 256, 0, stream>>>(X16, real_nodes, wbuf, batch, vnbuf, wihH,
                                     whhH, bih, bhh, out);
}
